// InvariantPointAttention_80169859547204
// MI455X (gfx1250) — matrix operand order repaired and checked by reading, not compiled, not run on hardware
//
#include <hip/hip_runtime.h>
#include <math.h>

// Invariant Point Attention, fused single-pass over `pair` (268 MB read once
// -> ~11.5us HBM floor at 23.3 TB/s). bf16 WMMA (16x16x32) for QK^T, point
// dot, P@V, P@Vg; f32 WMMA (16x16x4) for the 1024x448 @ 448x128 output GEMM.
// pair tiles are streamed HBM -> LDS with GLOBAL_LOAD_ASYNC_TO_LDS_B128
// (ASYNCcnt), double-buffered so tile jb+1 transfers while tile jb computes.

#define LSEQ   1024
#define NH     4
#define DHD    32
#define DMODEL 128
#define DPAIR  64
#define NVP    4
#define TI     16
#define TJ     32
#define CWIDTH 448   // 128 (o_s) + 256 (o_pair) + 48 (o_pt) + 16 (o_pt_n)

typedef __attribute__((ext_vector_type(16))) __bf16 v16bf;
typedef __attribute__((ext_vector_type(8)))  float  v8f;
typedef __attribute__((ext_vector_type(2)))  float  v2f;

static __device__ __forceinline__ unsigned short f2bf(float f) {
  unsigned int u = __float_as_uint(f);
  u += 0x7fffu + ((u >> 16) & 1u);              // round-to-nearest-even
  return (unsigned short)(u >> 16);
}
static __device__ __forceinline__ float bf2f(unsigned short s) {
  return __uint_as_float(((unsigned int)s) << 16);
}

// A operand 16x32 bf16: lane m=L%16, hi=L/16 holds
// K = 16*hi .. 16*hi+15 of row m (contiguous, matches ldB).
static __device__ __forceinline__ v16bf ldA(const unsigned short* row, int hi) {
  union { unsigned short u[16]; v16bf v; } t;
#pragma unroll
  for (int e = 0; e < 8; ++e) t.u[e] = row[hi * 16 + e];
#pragma unroll
  for (int e = 0; e < 8; ++e) t.u[8 + e] = row[hi * 16 + 8 + e];
  return t.v;
}
// B operand 32x16 bf16: lane n=L%16, hi=L/16 holds K = 16*hi .. 16*hi+15 of
// column n; caller passes pointer to those 16 contiguous bf16.
static __device__ __forceinline__ v16bf ldB(const unsigned short* k16) {
  union { unsigned short u[16]; v16bf v; } t;
#pragma unroll
  for (int e = 0; e < 16; ++e) t.u[e] = k16[e];
  return t.v;
}
static __device__ __forceinline__ v8f wmma_bf16(v16bf a, v16bf b, v8f c) {
  return __builtin_amdgcn_wmma_f32_16x16x32_bf16(false, a, false, b, (short)0, c,
                                                 false, false);
}

// CDNA5 async DMA: copy 16 bytes/lane from global to LDS, tracked by ASYNCcnt.
static __device__ __forceinline__ void async_copy_b128(unsigned lds_off,
                                                       const void* gaddr) {
  const unsigned long long ga = (unsigned long long)(uintptr_t)gaddr;
  asm volatile("global_load_async_to_lds_b128 %0, %1, off"
               :: "v"(lds_off), "v"(ga) : "memory");
}
static __device__ __forceinline__ void wait_async0() {
  asm volatile("s_wait_asynccnt 0" ::: "memory");
}

// ---------------------------------------------------------------------------
// Kernel 1: LayerNorm + projections + rigid transforms (one block per residue)
// ---------------------------------------------------------------------------
__global__ void __launch_bounds__(128)
ipa_prep(const float* __restrict__ single, const float* __restrict__ T,
         const float* __restrict__ ln_g, const float* __restrict__ ln_b,
         const float* __restrict__ Wq, const float* __restrict__ Wk,
         const float* __restrict__ Wv, const float* __restrict__ Wqpt,
         const float* __restrict__ Wkpt, const float* __restrict__ Wvpt,
         unsigned short* __restrict__ Qh, unsigned short* __restrict__ Kh,
         unsigned short* __restrict__ Qg, unsigned short* __restrict__ Kg,
         unsigned short* __restrict__ Vt, unsigned short* __restrict__ Vgt,
         float* __restrict__ qn, float* __restrict__ kn) {
  __shared__ float xL[DMODEL], zL[DMODEL];
  __shared__ float qptL[48], kptL[48], vptL[48], qgL[48], kgL[48];
  const int i = blockIdx.x;
  const int t = threadIdx.x;

  xL[t] = single[(size_t)i * DMODEL + t];
  __syncthreads();
  float s1 = 0.f, s2 = 0.f;
#pragma unroll 8
  for (int k = 0; k < DMODEL; ++k) { float v = xL[k]; s1 += v; s2 += v * v; }
  const float mu = s1 * (1.f / DMODEL);
  const float var = s2 * (1.f / DMODEL) - mu * mu;
  const float rs = rsqrtf(var + 1e-5f);
  zL[t] = (xL[t] - mu) * rs * ln_g[t] + ln_b[t];
  __syncthreads();

  // scalar/vector projections (small: 1024 x 528 x 128 MACs total)
  float qd = 0.f, kd = 0.f, vd = 0.f;
#pragma unroll 4
  for (int k = 0; k < DMODEL; ++k) {
    const float zv = zL[k];
    qd += zv * Wq[k * DMODEL + t];
    kd += zv * Wk[k * DMODEL + t];
    vd += zv * Wv[k * DMODEL + t];
  }
  const int h = t >> 5, d = t & 31;
  Qh[((size_t)i * NH + h) * DHD + d] = f2bf(qd);
  Kh[((size_t)i * NH + h) * DHD + d] = f2bf(kd);
  Vt[(size_t)t * LSEQ + i] = f2bf(vd);  // Vt[h][d][i], t = h*32+d
  if (t < 48) {
    float qp = 0.f, kp = 0.f, vp = 0.f;
#pragma unroll 4
    for (int k = 0; k < DMODEL; ++k) {
      const float zv = zL[k];
      qp += zv * Wqpt[k * 48 + t];
      kp += zv * Wkpt[k * 48 + t];
      vp += zv * Wvpt[k * 48 + t];
    }
    qptL[t] = qp; kptL[t] = kp; vptL[t] = vp;
  }
  __syncthreads();

  // zero-pad the 12-wide point-dot operands out to K=32 / 16 rows
  if (d >= 12) {
    Qg[((size_t)i * NH + h) * DHD + d] = 0;
    Kg[((size_t)i * NH + h) * DHD + d] = 0;
  }
  if (t < 64) {
    const int hh = t >> 4, r = t & 15;
    if (r >= 12) Vgt[((size_t)(hh * 16) + r) * LSEQ + i] = 0;
  }
  if (t < 16) {  // rigid transforms: one (head, point) per thread
    const int hh = t >> 2, p = t & 3;
    float R[3][3], tr[3];
#pragma unroll
    for (int x = 0; x < 3; ++x) {
#pragma unroll
      for (int y = 0; y < 3; ++y) R[x][y] = T[(size_t)i * 16 + x * 4 + y];
      tr[x] = T[(size_t)i * 16 + x * 4 + 3];
    }
    const float* qv = &qptL[hh * 12 + p * 3];
    const float* kv = &kptL[hh * 12 + p * 3];
    const float* vv = &vptL[hh * 12 + p * 3];
#pragma unroll
    for (int x = 0; x < 3; ++x) {
      const float qg_ = tr[x] + R[x][0] * qv[0] + R[x][1] * qv[1] + R[x][2] * qv[2];
      const float kg_ = tr[x] + R[x][0] * kv[0] + R[x][1] * kv[1] + R[x][2] * kv[2];
      const float vg_ = tr[x] + R[x][0] * vv[0] + R[x][1] * vv[1] + R[x][2] * vv[2];
      Qg[((size_t)i * NH + hh) * DHD + p * 3 + x] = f2bf(qg_);
      Kg[((size_t)i * NH + hh) * DHD + p * 3 + x] = f2bf(kg_);
      Vgt[((size_t)(hh * 16) + p * 3 + x) * LSEQ + i] = f2bf(vg_);
      qgL[hh * 12 + p * 3 + x] = qg_;
      kgL[hh * 12 + p * 3 + x] = kg_;
    }
  }
  __syncthreads();
  if (t < NH) {  // squared norms for d2 = |Q|^2 + |K|^2 - 2 Q.K
    float sq = 0.f, sk = 0.f;
#pragma unroll
    for (int r = 0; r < 12; ++r) {
      const float a = qgL[t * 12 + r]; sq += a * a;
      const float b = kgL[t * 12 + r]; sk += b * b;
    }
    qn[(size_t)i * NH + t] = sq;
    kn[(size_t)i * NH + t] = sk;
  }
}

// ---------------------------------------------------------------------------
// Kernel 2: fused IPA attention. One block = 16 query rows, 4 waves (1/head).
// Streams pair[] once via double-buffered async HBM->LDS DMA + online softmax.
// ---------------------------------------------------------------------------
#define SM_PAIR0 0
#define SM_PAIR1 (SM_PAIR0 + TI * TJ * DPAIR * 4)     // +131072
#define SM_B     (SM_PAIR1 + TI * TJ * DPAIR * 4)     // +131072 = 262144
#define SM_P     (SM_B + NH * TI * TJ * 4)            // +8192
#define SM_WB    (SM_P + NH * TI * TJ * 2)            // +4096
#define SM_SC    (SM_WB + DPAIR * NH * 4)             // +1024
#define SM_OG    (SM_SC + NH * TI * 4)                // +256
#define SM_TOT   (SM_OG + NH * TI * 16 * 4)           // +4096 => 279808 B

__global__ void __launch_bounds__(128)
ipa_attn(const float* __restrict__ pair, const float* __restrict__ T,
         const float* __restrict__ w_C, const float* __restrict__ Wb,
         const unsigned short* __restrict__ Qh, const unsigned short* __restrict__ Kh,
         const unsigned short* __restrict__ Qg, const unsigned short* __restrict__ Kg,
         const unsigned short* __restrict__ Vt, const unsigned short* __restrict__ Vgt,
         const float* __restrict__ qn, const float* __restrict__ kn,
         float* __restrict__ concat) {
  extern __shared__ char smem[];
  // NOTE: no pointer arrays derived from the LDS base (lld cannot relocate a
  // constant aggregate holding an addrspacecast of @smem) — buffer selection
  // is done with arithmetic at each use site instead.
  float* bL = (float*)(smem + SM_B);                      // [4][16][32]
  unsigned short* pL = (unsigned short*)(smem + SM_P);    // [4][16][32] bf16
  float* wbL = (float*)(smem + SM_WB);                    // [64][4]
  float* scL = (float*)(smem + SM_SC);                    // [4][16]
  float* ogL = (float*)(smem + SM_OG);                    // [4][16][16]

  const int tid = threadIdx.x;
  const int h = tid >> 5;        // wave == head
  const int lane = tid & 31;
  const int ln16 = lane & 15;
  const int hi = lane >> 4;
  const int I0 = blockIdx.x * TI;

  for (int k = tid; k < DPAIR * NH; k += 128) wbL[k] = Wb[k];

  // issue async DMA of one 16x32x64 fp32 pair tile into an LDS buffer:
  // 8192 B128 transfers, 64 per thread (ASYNCcnt saturates at 63 and the
  // hardware back-pressures issue, which is fine).
  const unsigned ldsStatic = __builtin_amdgcn_groupstaticsize();
  auto issue_tile = [&](int jbN, unsigned bufOff) {
    const int J0n = jbN * TJ;
    const unsigned base = ldsStatic + bufOff;
#pragma unroll 4
    for (int k = 0; k < 64; ++k) {
      const int tIdx = tid + k * 128;
      const int row = tIdx >> 4;              // (il,jl) flattened, 0..511
      const int c4 = (tIdx & 15) << 2;        // float offset 0..60
      const int il = row >> 5, jl = row & 31;
      const float* src = pair + (((size_t)(I0 + il)) * LSEQ + (J0n + jl)) * DPAIR + c4;
      async_copy_b128(base + (unsigned)((row * DPAIR + c4) * 4), src);
    }
  };

  const float wsp = log1pf(__expf(w_C[h]));      // softplus(w_C[h])
  const float scaleA = 0.17677669529663687f;     // DH^-0.5
  const v8f vz = {0.f, 0.f, 0.f, 0.f, 0.f, 0.f, 0.f, 0.f};

  const v16bf a_q = ldA(Qh + (((size_t)(I0 + ln16)) * NH + h) * DHD, hi);
  const v16bf a_g = ldA(Qg + (((size_t)(I0 + ln16)) * NH + h) * DHD, hi);
  float qn8[8];
#pragma unroll
  for (int r = 0; r < 8; ++r) qn8[r] = qn[(size_t)(I0 + r + 8 * hi) * NH + h];

  float mrow[8], lrow[8];
#pragma unroll
  for (int r = 0; r < 8; ++r) { mrow[r] = -1e30f; lrow[r] = 0.f; }
  v8f c0 = vz, c1 = vz, cg = vz;
  float opair[32];  // this lane owns o_pair channels {2*lane, 2*lane+1} x 16 rows
#pragma unroll
  for (int k = 0; k < 32; ++k) opair[k] = 0.f;

  issue_tile(0, (unsigned)SM_PAIR0);  // prologue: start tile 0 DMA

  for (int jb = 0; jb < LSEQ / TJ; ++jb) {
    const int J0 = jb * TJ;
    const unsigned curOff = (jb & 1) ? (unsigned)SM_PAIR1 : (unsigned)SM_PAIR0;
    const float* pcf = (const float*)(smem + curOff);
    wait_async0();       // my wave's DMA for tile jb complete
    __syncthreads();     // all waves' DMA done; previous compute finished
    if (jb + 1 < LSEQ / TJ)
      issue_tile(jb + 1, ((jb + 1) & 1) ? (unsigned)SM_PAIR1 : (unsigned)SM_PAIR0);

    // pair bias for all 4 heads (cooperative, fp32 from LDS)
    for (int r = tid; r < TI * TJ; r += 128) {
      const float* pr = pcf + r * DPAIR;
      float b0 = 0.f, b1 = 0.f, b2 = 0.f, b3 = 0.f;
#pragma unroll 8
      for (int c = 0; c < DPAIR; ++c) {
        const float pv = pr[c];
        b0 += pv * wbL[c * NH + 0]; b1 += pv * wbL[c * NH + 1];
        b2 += pv * wbL[c * NH + 2]; b3 += pv * wbL[c * NH + 3];
      }
      const int il = r >> 5, jl = r & 31;
      bL[(0 * TI + il) * TJ + jl] = b0; bL[(1 * TI + il) * TJ + jl] = b1;
      bL[(2 * TI + il) * TJ + jl] = b2; bL[(3 * TI + il) * TJ + jl] = b3;
    }
    __syncthreads();

    // logits via WMMA: S = Q.K^T (K=32=DH), G = Qg.Kg^T (12 dims zero-padded)
    float lg[2][8];
#pragma unroll
    for (int sub = 0; sub < 2; ++sub) {
      const int jn = J0 + sub * 16 + ln16;
      const v16bf bk = ldB(Kh + ((size_t)jn * NH + h) * DHD + hi * 16);
      const v16bf bg = ldB(Kg + ((size_t)jn * NH + h) * DHD + hi * 16);
      const v8f s = wmma_bf16(a_q, bk, vz);
      const v8f g = wmma_bf16(a_g, bg, vz);
      const float knj = kn[(size_t)jn * NH + h];
#pragma unroll
      for (int r = 0; r < 8; ++r) {
        const float bia = bL[(h * TI + r + 8 * hi) * TJ + sub * 16 + ln16];
        lg[sub][r] = s[r] * scaleA + bia - 0.5f * wsp * (qn8[r] + knj - 2.0f * g[r]);
      }
    }
    // online softmax (row reductions stay inside each 16-lane half)
    float sc[8];
#pragma unroll
    for (int r = 0; r < 8; ++r) {
      float mx = fmaxf(lg[0][r], lg[1][r]);
#pragma unroll
      for (int msk = 1; msk < 16; msk <<= 1) mx = fmaxf(mx, __shfl_xor(mx, msk, 32));
      const float mn = fmaxf(mrow[r], mx);
      sc[r] = __expf(mrow[r] - mn);
      mrow[r] = mn;
      lg[0][r] = __expf(lg[0][r] - mn);
      lg[1][r] = __expf(lg[1][r] - mn);
      float ps = lg[0][r] + lg[1][r];
#pragma unroll
      for (int msk = 1; msk < 16; msk <<= 1) ps += __shfl_xor(ps, msk, 32);
      lrow[r] = lrow[r] * sc[r] + ps;
      c0[r] *= sc[r]; c1[r] *= sc[r]; cg[r] *= sc[r];
    }
    // broadcast per-row rescale factors for the o_pair accumulator
    if (ln16 == 0) {
#pragma unroll
      for (int r = 0; r < 8; ++r) scL[h * TI + 8 * hi + r] = sc[r];
    }
#pragma unroll
    for (int il = 0; il < 16; ++il) {
      const float f = scL[h * TI + il];
      opair[il * 2] *= f; opair[il * 2 + 1] *= f;
    }
    // write P tile (bf16) and reload in A layout for the PV WMMAs
#pragma unroll
    for (int r = 0; r < 8; ++r) {
      pL[(h * TI + r + 8 * hi) * TJ + ln16] = f2bf(lg[0][r]);
      pL[(h * TI + r + 8 * hi) * TJ + 16 + ln16] = f2bf(lg[1][r]);
    }
    const v16bf a_p = ldA(pL + (h * TI + ln16) * TJ, hi);
    const v16bf bv0 = ldB(Vt + ((size_t)(h * DHD) + ln16) * LSEQ + J0 + hi * 16);
    const v16bf bv1 = ldB(Vt + ((size_t)(h * DHD) + 16 + ln16) * LSEQ + J0 + hi * 16);
    const v16bf bvg = ldB(Vgt + ((size_t)(h * 16) + ln16) * LSEQ + J0 + hi * 16);
    c0 = wmma_bf16(a_p, bv0, c0);   // o_s d 0..15
    c1 = wmma_bf16(a_p, bv1, c1);   // o_s d 16..31
    cg = wmma_bf16(a_p, bvg, cg);   // o_pt_g (12 valid cols)
    // o_pair accumulation (lane owns 2 channels; p broadcast via LDS)
#pragma unroll 2
    for (int il = 0; il < 16; ++il) {
      float a0 = opair[il * 2], a1 = opair[il * 2 + 1];
      const unsigned short* prow = pL + (h * TI + il) * TJ;
      const float* prb = pcf + il * TJ * DPAIR + lane * 2;
#pragma unroll 4
      for (int jl = 0; jl < TJ; ++jl) {
        const float pv = bf2f(prow[jl]);
        const float2 w2 = *(const float2*)(prb + jl * DPAIR);
        a0 += pv * w2.x; a1 += pv * w2.y;
      }
      opair[il * 2] = a0; opair[il * 2 + 1] = a1;
    }
  }

  // finalize: divide by softmax denominator, inverse rigid transform, norms
  float inv[8];
#pragma unroll
  for (int r = 0; r < 8; ++r) {
    inv[r] = 1.0f / lrow[r];
    c0[r] *= inv[r]; c1[r] *= inv[r]; cg[r] *= inv[r];
  }
#pragma unroll
  for (int r = 0; r < 8; ++r) {
    const int row = I0 + r + 8 * hi;
    concat[(size_t)row * CWIDTH + h * DHD + ln16] = c0[r];
    concat[(size_t)row * CWIDTH + h * DHD + 16 + ln16] = c1[r];
    ogL[(h * TI + r + 8 * hi) * 16 + ln16] = cg[r];
  }
  if (ln16 == 0) {
#pragma unroll
    for (int r = 0; r < 8; ++r) scL[h * TI + 8 * hi + r] = inv[r];
  }
#pragma unroll
  for (int il = 0; il < 16; ++il) {
    const float f = scL[h * TI + il];
    concat[(size_t)(I0 + il) * CWIDTH + 128 + h * DPAIR + lane * 2] = opair[il * 2] * f;
    concat[(size_t)(I0 + il) * CWIDTH + 128 + h * DPAIR + lane * 2 + 1] = opair[il * 2 + 1] * f;
  }
  __syncthreads();
  for (int tsk = lane; tsk < 64; tsk += 32) {
    const int mm = tsk >> 2, p = tsk & 3;
    const int i = I0 + mm;
    float dx[3];
#pragma unroll
    for (int y = 0; y < 3; ++y)
      dx[y] = ogL[(h * TI + mm) * 16 + p * 3 + y] - T[(size_t)i * 16 + y * 4 + 3];
    float o[3];
#pragma unroll
    for (int x = 0; x < 3; ++x) {
      o[x] = T[(size_t)i * 16 + 0 * 4 + x] * dx[0] + T[(size_t)i * 16 + 1 * 4 + x] * dx[1] +
             T[(size_t)i * 16 + 2 * 4 + x] * dx[2];
      concat[(size_t)i * CWIDTH + 384 + h * 12 + p * 3 + x] = o[x];
    }
    concat[(size_t)i * CWIDTH + 432 + h * NVP + p] =
        sqrtf(o[0] * o[0] + o[1] * o[1] + o[2] * o[2]);
  }
}

// ---------------------------------------------------------------------------
// Kernel 3: out = single + concat @ Wout + b_out  (fp32 WMMA 16x16x4)
// ---------------------------------------------------------------------------
__global__ void __launch_bounds__(256)
ipa_out(const float* __restrict__ concat, const float* __restrict__ single,
        const float* __restrict__ Wout, const float* __restrict__ b_out,
        float* __restrict__ out) {
  const int tid = threadIdx.x;
  const int w = tid >> 5, lane = tid & 31, ln16 = lane & 15, hi = lane >> 4;
  const int I0 = blockIdx.x * 16;
  const int n0 = w * 16;
#if __has_builtin(__builtin_amdgcn_wmma_f32_16x16x4_f32)
  v8f c = {0.f, 0.f, 0.f, 0.f, 0.f, 0.f, 0.f, 0.f};
  for (int kk = 0; kk < CWIDTH / 4; ++kk) {
    const int k0 = kk * 4 + hi * 2;  // 32-bit A/B layout: lane holds K = {2hi, 2hi+1}
    v2f a, b;
    a.x = concat[(size_t)(I0 + ln16) * CWIDTH + k0];
    a.y = concat[(size_t)(I0 + ln16) * CWIDTH + k0 + 1];
    b.x = Wout[(size_t)k0 * DMODEL + n0 + ln16];
    b.y = Wout[(size_t)(k0 + 1) * DMODEL + n0 + ln16];
    c = __builtin_amdgcn_wmma_f32_16x16x4_f32(false, a, false, b, (short)0, c, false, false);
  }
#pragma unroll
  for (int r = 0; r < 8; ++r) {
    const int row = I0 + r + 8 * hi;
    const int col = n0 + ln16;
    out[(size_t)row * DMODEL + col] = single[(size_t)row * DMODEL + col] + b_out[col] + c[r];
  }
#else
  (void)w; (void)ln16; (void)hi; (void)n0;
  for (int e = tid; e < 16 * DMODEL; e += 256) {
    const int mm = e >> 7, col = e & 127;
    const int row = I0 + mm;
    float acc = b_out[col] + single[(size_t)row * DMODEL + col];
    for (int k = 0; k < CWIDTH; ++k)
      acc += concat[(size_t)row * CWIDTH + k] * Wout[(size_t)k * DMODEL + col];
    out[(size_t)row * DMODEL + col] = acc;
  }
#endif
}

// ---------------------------------------------------------------------------
extern "C" void kernel_launch(void* const* d_in, const int* in_sizes, int n_in,
                              void* d_out, int out_size, void* d_ws, size_t ws_size,
                              hipStream_t stream) {
  const float* single = (const float*)d_in[0];
  const float* pair   = (const float*)d_in[1];
  const float* T      = (const float*)d_in[2];
  const float* w_C    = (const float*)d_in[3];
  const float* ln_g   = (const float*)d_in[4];
  const float* ln_b   = (const float*)d_in[5];
  const float* Wq     = (const float*)d_in[6];
  const float* Wk     = (const float*)d_in[7];
  const float* Wv     = (const float*)d_in[8];
  const float* Wqpt   = (const float*)d_in[9];
  const float* Wkpt   = (const float*)d_in[10];
  const float* Wvpt   = (const float*)d_in[11];
  const float* Wb     = (const float*)d_in[12];
  const float* Wout   = (const float*)d_in[13];
  const float* b_out  = (const float*)d_in[14];
  float* out = (float*)d_out;

  char* ws = (char*)d_ws;
  size_t off = 0;
  auto carve = [&](size_t bytes) {
    char* p = ws + off;
    off += (bytes + 255) & ~(size_t)255;
    return p;
  };
  unsigned short* Qh  = (unsigned short*)carve((size_t)LSEQ * NH * DHD * 2);
  unsigned short* Kh  = (unsigned short*)carve((size_t)LSEQ * NH * DHD * 2);
  unsigned short* Qg  = (unsigned short*)carve((size_t)LSEQ * NH * DHD * 2);
  unsigned short* Kg  = (unsigned short*)carve((size_t)LSEQ * NH * DHD * 2);
  unsigned short* Vt  = (unsigned short*)carve((size_t)NH * DHD * LSEQ * 2);
  unsigned short* Vgt = (unsigned short*)carve((size_t)NH * 16 * LSEQ * 2);
  float* qn     = (float*)carve((size_t)LSEQ * NH * 4);
  float* kn     = (float*)carve((size_t)LSEQ * NH * 4);
  float* concat = (float*)carve((size_t)LSEQ * CWIDTH * 4);
  (void)in_sizes; (void)n_in; (void)out_size; (void)ws_size;

  ipa_prep<<<LSEQ, 128, 0, stream>>>(single, T, ln_g, ln_b, Wq, Wk, Wv, Wqpt, Wkpt,
                                     Wvpt, Qh, Kh, Qg, Kg, Vt, Vgt, qn, kn);
  ipa_attn<<<LSEQ / TI, 128, SM_TOT, stream>>>(pair, T, w_C, Wb, Qh, Kh, Qg, Kg, Vt,
                                               Vgt, qn, kn, concat);
  ipa_out<<<LSEQ / 16, 256, 0, stream>>>(concat, single, Wout, b_out, out);
}
